// GroupedQueryAttention_31078383354634
// MI455X (gfx1250) — matrix operand order repaired and checked by reading, not compiled, not run on hardware
//
#include <hip/hip_runtime.h>
#include <hip/hip_bf16.h>

#define D_IN      2048
#define D_OUTV    2048
#define NUM_HEADS 32
#define NUM_KV    8
#define HEAD_DIM  64
#define BB        2
#define SS        2048
#define NTOK      (BB*SS)            // 4096
#define KV_COLS   (NUM_KV*HEAD_DIM)  // 512

typedef __attribute__((ext_vector_type(16))) _Float16 v16h;
typedef __attribute__((ext_vector_type(8)))  _Float16 v8h;
typedef __attribute__((ext_vector_type(8)))  float    v8f;
typedef __attribute__((ext_vector_type(4)))  int      v4i;

// ---- gfx1250 async global->LDS path (ASYNCcnt), with safe fallback --------
#if defined(__has_builtin)
#  if __has_builtin(__builtin_amdgcn_global_load_async_to_lds_b128)
#    define HAS_ASYNC_LDS 1
#  endif
#endif
#ifndef HAS_ASYNC_LDS
#  define HAS_ASYNC_LDS 0
#endif

#if HAS_ASYNC_LDS
typedef __attribute__((address_space(1))) void as1_void;
typedef __attribute__((address_space(3))) void as3_void;
typedef __attribute__((address_space(1))) v4i  as1_v4i;
typedef __attribute__((address_space(3))) v4i  as3_v4i;
#endif

static __device__ __forceinline__ v16h cat16(v8h lo, v8h hi) {
  v16h r;
#pragma unroll
  for (int i = 0; i < 8; ++i) { r[i] = lo[i]; r[8 + i] = hi[i]; }
  return r;
}

// A fragment (16x32 f16, MxK). Layout: lanes 0-15 hold K [0..16),
// lanes 16-31 hold K [16..32). src row-major, row stride ld (halfs).
static __device__ __forceinline__ v16h loadA(const _Float16* base, int row, int kk,
                                             int ld, int lane) {
  const _Float16* p = base + (size_t)row * ld + kk + ((lane >> 4) * 16);
  v8h lo = *(const v8h*)p;
  v8h hi = *(const v8h*)(p + 8);
  return cat16(lo, hi);
}

// B fragment (32x16 f16, KxN). Layout used here (same k order as loadA): lane holds column n=lane&15,
// lanes 0-15 k=[0..16), lanes 16-31 k=[16..32). src is [n][k], row stride ld.
static __device__ __forceinline__ v16h loadB(const _Float16* base, int n, int kk,
                                             int ld, int lane) {
  const _Float16* p = base + (size_t)n * ld + kk + ((lane >> 4) * 16);
  v8h lo = *(const v8h*)p;
  v8h hi = *(const v8h*)(p + 8);
  return cat16(lo, hi);
}

static __device__ __forceinline__ v8f wmma_f16(v16h a, v16h b, v8f c) {
  return __builtin_amdgcn_wmma_f32_16x16x32_f16(false, a, false, b, (short)0, c,
                                                false, false);
}

// Stage a 32-row x 32-k f16 A-slice into LDS: 128 threads x 16B each.
static __device__ __forceinline__ void stageA(const _Float16* gsrc, int gld,
                                              _Float16* ldst, int tid) {
  int r = tid >> 2;          // 0..31
  int c = (tid & 3) * 8;     // 0,8,16,24 halfs
  const _Float16* gp = gsrc + (size_t)r * gld + c;
  _Float16* lp = ldst + r * 32 + c;
#if HAS_ASYNC_LDS
  __builtin_amdgcn_global_load_async_to_lds_b128(
      (as1_v4i*)(as1_void*)gp, (as3_v4i*)(as3_void*)lp, 0, 0);
#else
  *(v8h*)lp = *(const v8h*)gp;
#endif
}

#if HAS_ASYNC_LDS
#define WAIT_ASYNC_1() asm volatile("s_wait_asynccnt 1" ::: "memory")
#define WAIT_ASYNC_0() asm volatile("s_wait_asynccnt 0" ::: "memory")
#else
#define WAIT_ASYNC_1()
#define WAIT_ASYNC_0()
#endif

// ---------------------------------------------------------------- converts
__global__ void cvt_f32_f16(const float* __restrict__ src,
                            _Float16* __restrict__ dst, int n) {
  int i = blockIdx.x * blockDim.x + threadIdx.x;
  if (i < n) dst[i] = (_Float16)src[i];
}

// src [K][N] f32 -> dst [N][K] f16 (transposed for B-fragment loads)
__global__ void cvt_w_t(const float* __restrict__ src,
                        _Float16* __restrict__ dst, int K, int N) {
  int i = blockIdx.x * blockDim.x + threadIdx.x;
  if (i < K * N) {
    int n = i / K, k = i % K;
    dst[(size_t)n * K + k] = (_Float16)src[(size_t)k * N + n];
  }
}

// ------------------------------------------------- fused QKV GEMM + RoPE
// Block = 4 waves sharing a 32-row A strip (double-buffered async LDS stage).
// Each wave owns one 64-col "unit" (a whole head): 32 Q + 8 K + 8 V = 48 units.
// Wave tile: 32x64 -> 8 accumulators, 8 WMMA per 32-k step.
__global__ __launch_bounds__(128) void qkv_rope_kernel(
    const _Float16* __restrict__ xh,    // [NTOK][D_IN]
    const _Float16* __restrict__ wqh,   // [D_OUTV][D_IN] transposed
    const _Float16* __restrict__ wkh,   // [KV_COLS][D_IN]
    const _Float16* __restrict__ wvh,   // [KV_COLS][D_IN]
    const float* __restrict__ cosT,     // [SS][64]
    const float* __restrict__ sinT,     // [SS][64]
    _Float16* __restrict__ Qr,          // [B][H][S][64]
    _Float16* __restrict__ Kr,          // [B][KV][S][64]
    _Float16* __restrict__ Vt)          // [B][KV][64][S]
{
  __shared__ __align__(16) _Float16 Ab[2][32][32];
  int tid = threadIdx.x, lane = tid & 31, w = tid >> 5;
  int strip = blockIdx.x / 12;   // 128 strips of 32 rows
  int ub    = blockIdx.x % 12;   // 12 unit-groups of 4
  int unit  = ub * 4 + w;
  int m0 = strip * 32;
  int half = lane >> 4, col = lane & 15;

  int kind, hd;
  const _Float16* wt;
  if (unit < 32)      { kind = 0; hd = unit;      wt = wqh; }
  else if (unit < 40) { kind = 1; hd = unit - 32; wt = wkh; }
  else                { kind = 2; hd = unit - 40; wt = wvh; }
  int cb = hd * HEAD_DIM;

  v8f acc[2][4];
#pragma unroll
  for (int r = 0; r < 2; ++r)
#pragma unroll
    for (int c = 0; c < 4; ++c) acc[r][c] = (v8f){};

  const _Float16* asrc = xh + (size_t)m0 * D_IN;
  stageA(asrc, D_IN, &Ab[0][0][0], tid);
  int cur = 0;
  for (int kk = 0; kk < D_IN; kk += 32) {
    bool more = (kk + 32) < D_IN;
    if (more) {
      stageA(asrc + kk + 32, D_IN, &Ab[cur ^ 1][0][0], tid);
      WAIT_ASYNC_1();
    } else {
      WAIT_ASYNC_0();
    }
    __syncthreads();
    v16h a0 = loadA(&Ab[cur][0][0], col,      0, 32, lane);
    v16h a1 = loadA(&Ab[cur][0][0], 16 + col, 0, 32, lane);
#pragma unroll
    for (int c = 0; c < 4; ++c) {
      v16h b = loadB(wt, cb + c * 16 + col, kk, D_IN, lane);
      acc[0][c] = wmma_f16(a0, b, acc[0][c]);
      acc[1][c] = wmma_f16(a1, b, acc[1][c]);
    }
    __syncthreads();
    cur ^= 1;
  }

  // ---- epilogue: rope pairs are (c,c+2): in-head dims dd and dd+32
  int b = m0 / SS;
  if (kind == 2) {
    size_t vbase = ((size_t)b * NUM_KV + hd) * HEAD_DIM;
#pragma unroll
    for (int r = 0; r < 2; ++r) {
      int s0 = (m0 % SS) + r * 16 + half * 8;
#pragma unroll
      for (int p = 0; p < 2; ++p) {
        int ddl = p * 16 + col;
        v8h pk_lo, pk_hi;
#pragma unroll
        for (int v = 0; v < 8; ++v) {
          pk_lo[v] = (_Float16)acc[r][p][v];
          pk_hi[v] = (_Float16)acc[r][p + 2][v];
        }
        *(v8h*)(Vt + (vbase + ddl)      * SS + s0) = pk_lo;
        *(v8h*)(Vt + (vbase + ddl + 32) * SS + s0) = pk_hi;
      }
    }
  } else {
    _Float16* dst = (kind == 0) ? Qr : Kr;
    int nh = (kind == 0) ? NUM_HEADS : NUM_KV;
#pragma unroll
    for (int r = 0; r < 2; ++r) {
#pragma unroll
      for (int p = 0; p < 2; ++p) {
        int ddl = p * 16 + col;
#pragma unroll
        for (int v = 0; v < 8; ++v) {
          int tok = m0 + r * 16 + v + half * 8;
          int s = tok % SS;
          float c  = cosT[s * 64 + ddl];  // angles duplicated: cos[dd]==cos[dd+32]
          float sn = sinT[s * 64 + ddl];
          float lo = acc[r][p][v], hi = acc[r][p + 2][v];
          size_t rb = (((size_t)b * nh + hd) * SS + s) * HEAD_DIM;
          dst[rb + ddl]      = (_Float16)(lo * c - hi * sn);
          dst[rb + ddl + 32] = (_Float16)(hi * c + lo * sn);
        }
      }
    }
  }
}

// --------------------------------------------------- causal flash attention
__global__ __launch_bounds__(128) void attn_kernel(
    const _Float16* __restrict__ Qr,    // [B][H][S][64]
    const _Float16* __restrict__ Kr,    // [B][KV][S][64]
    const _Float16* __restrict__ Vt,    // [B][KV][64][S]
    _Float16* __restrict__ ctxh)        // [NTOK][2048]
{
  __shared__ __align__(16) _Float16 Pl[4][16][32];
  int tid = threadIdx.x, lane = tid & 31, w = tid >> 5;
  int wave = blockIdx.x * 4 + w;
  int qt = wave & 127;          // S/16
  int h  = (wave >> 7) & 31;
  int b  = wave >> 12;
  int g  = h >> 2;              // NUM_REPEAT = 4
  int q0 = qt * 16;
  int half = lane >> 4, col = lane & 15;

  const _Float16* Qb = Qr + ((size_t)b * NUM_HEADS + h) * SS * HEAD_DIM;
  const _Float16* Kb = Kr + ((size_t)b * NUM_KV + g) * SS * HEAD_DIM;
  const _Float16* Vb = Vt + ((size_t)b * NUM_KV + g) * HEAD_DIM * SS;

  v16h qa0 = loadA(Qb, q0 + col, 0,  HEAD_DIM, lane);
  v16h qa1 = loadA(Qb, q0 + col, 32, HEAD_DIM, lane);

  v8f ctx0 = {}, ctx1 = {}, ctx2 = {}, ctx3 = {};
  float mrow[8], lrow[8];
#pragma unroll
  for (int v = 0; v < 8; ++v) { mrow[v] = -3.0e38f; lrow[v] = 0.f; }
  const float scl = 0.125f;  // HEAD_DIM^-0.5

  for (int kc = 0; kc < q0 + 16; kc += 32) {
    v8f s0 = {}, s1 = {};
    s0 = wmma_f16(qa0, loadB(Kb, kc + col,      0,  HEAD_DIM, lane), s0);
    s0 = wmma_f16(qa1, loadB(Kb, kc + col,      32, HEAD_DIM, lane), s0);
    s1 = wmma_f16(qa0, loadB(Kb, kc + 16 + col, 0,  HEAD_DIM, lane), s1);
    s1 = wmma_f16(qa1, loadB(Kb, kc + 16 + col, 32, HEAD_DIM, lane), s1);

    float corr[8];
#pragma unroll
    for (int v = 0; v < 8; ++v) {
      int row = q0 + v + half * 8;
      float x0 = s0[v] * scl; if (kc + col > row)      x0 = -3.0e38f;
      float x1 = s1[v] * scl; if (kc + 16 + col > row) x1 = -3.0e38f;

      float rm = fmaxf(x0, x1);
#pragma unroll
      for (int off = 1; off < 16; off <<= 1) rm = fmaxf(rm, __shfl_xor(rm, off, 32));
      float mn = fmaxf(mrow[v], rm);
      float c  = __expf(mrow[v] - mn);
      float p0 = __expf(x0 - mn);
      float p1 = __expf(x1 - mn);
      float rs = p0 + p1;
#pragma unroll
      for (int off = 1; off < 16; off <<= 1) rs += __shfl_xor(rs, off, 32);
      lrow[v] = lrow[v] * c + rs;
      mrow[v] = mn;
      corr[v] = c;
      int r = v + half * 8;
      Pl[w][r][col]      = (_Float16)p0;   // C-layout -> LDS transpose buffer
      Pl[w][r][16 + col] = (_Float16)p1;
    }
#pragma unroll
    for (int v = 0; v < 8; ++v) {
      ctx0[v] *= corr[v]; ctx1[v] *= corr[v];
      ctx2[v] *= corr[v]; ctx3[v] *= corr[v];
    }

    asm volatile("s_wait_dscnt 0" ::: "memory");  // LDS write->read, same wave

    const _Float16* pb = &Pl[w][0][0];
    v16h pa = loadA(pb, col, 0, 32, lane);
    ctx0 = wmma_f16(pa, loadB(Vb,  0 + col, kc, SS, lane), ctx0);
    ctx1 = wmma_f16(pa, loadB(Vb, 16 + col, kc, SS, lane), ctx1);
    ctx2 = wmma_f16(pa, loadB(Vb, 32 + col, kc, SS, lane), ctx2);
    ctx3 = wmma_f16(pa, loadB(Vb, 48 + col, kc, SS, lane), ctx3);
  }

#pragma unroll
  for (int v = 0; v < 8; ++v) {
    float inv = 1.0f / lrow[v];
    int tok = b * SS + q0 + v + half * 8;
    size_t rb = (size_t)tok * D_OUTV + h * HEAD_DIM;
    ctxh[rb + 0  + col] = (_Float16)(ctx0[v] * inv);
    ctxh[rb + 16 + col] = (_Float16)(ctx1[v] * inv);
    ctxh[rb + 32 + col] = (_Float16)(ctx2[v] * inv);
    ctxh[rb + 48 + col] = (_Float16)(ctx3[v] * inv);
  }
}

// ----------------------------------------------- output projection + bias
// Same blocked structure: 32x64 wave tiles, async-staged shared A strip.
__global__ __launch_bounds__(128) void out_proj_kernel(
    const _Float16* __restrict__ ctxh,  // [NTOK][2048]
    const _Float16* __restrict__ woh,   // [2048][2048] transposed
    const float* __restrict__ bo,
    float* __restrict__ out)            // [NTOK][2048] f32
{
  __shared__ __align__(16) _Float16 Ab[2][32][32];
  int tid = threadIdx.x, lane = tid & 31, w = tid >> 5;
  int strip = blockIdx.x >> 3;   // 128 strips of 32 rows
  int ub    = blockIdx.x & 7;    // 8 unit-groups of 4
  int unit  = ub * 4 + w;        // 32 units of 64 cols
  int m0 = strip * 32, cb = unit * 64;
  int half = lane >> 4, col = lane & 15;

  v8f acc[2][4];
#pragma unroll
  for (int r = 0; r < 2; ++r)
#pragma unroll
    for (int c = 0; c < 4; ++c) acc[r][c] = (v8f){};

  const _Float16* asrc = ctxh + (size_t)m0 * D_OUTV;
  stageA(asrc, D_OUTV, &Ab[0][0][0], tid);
  int cur = 0;
  for (int kk = 0; kk < D_OUTV; kk += 32) {
    bool more = (kk + 32) < D_OUTV;
    if (more) {
      stageA(asrc + kk + 32, D_OUTV, &Ab[cur ^ 1][0][0], tid);
      WAIT_ASYNC_1();
    } else {
      WAIT_ASYNC_0();
    }
    __syncthreads();
    v16h a0 = loadA(&Ab[cur][0][0], col,      0, 32, lane);
    v16h a1 = loadA(&Ab[cur][0][0], 16 + col, 0, 32, lane);
#pragma unroll
    for (int c = 0; c < 4; ++c) {
      v16h b = loadB(woh, cb + c * 16 + col, kk, D_OUTV, lane);
      acc[0][c] = wmma_f16(a0, b, acc[0][c]);
      acc[1][c] = wmma_f16(a1, b, acc[1][c]);
    }
    __syncthreads();
    cur ^= 1;
  }

#pragma unroll
  for (int c = 0; c < 4; ++c) {
    float bias = bo[cb + c * 16 + col];
#pragma unroll
    for (int r = 0; r < 2; ++r) {
#pragma unroll
      for (int v = 0; v < 8; ++v) {
        int row = m0 + r * 16 + v + half * 8;
        out[(size_t)row * D_OUTV + cb + c * 16 + col] = acc[r][c][v] + bias;
      }
    }
  }
}

// ------------------------------------------------------------------ launch
extern "C" void kernel_launch(void* const* d_in, const int* in_sizes, int n_in,
                              void* d_out, int out_size, void* d_ws, size_t ws_size,
                              hipStream_t stream) {
  const float* x    = (const float*)d_in[0];
  // d_in[1] = mask (unused; causal mask computed analytically)
  const float* cosT = (const float*)d_in[2];
  const float* sinT = (const float*)d_in[3];
  const float* wq   = (const float*)d_in[4];
  const float* wk   = (const float*)d_in[5];
  const float* wv   = (const float*)d_in[6];
  const float* wo   = (const float*)d_in[7];
  const float* bo   = (const float*)d_in[8];
  float* out = (float*)d_out;

  char* ws = (char*)d_ws;
  size_t off = 0;
  _Float16* xh   = (_Float16*)(ws + off); off += (size_t)NTOK * D_IN * 2;
  _Float16* wqh  = (_Float16*)(ws + off); off += (size_t)D_OUTV * D_IN * 2;
  _Float16* wkh  = (_Float16*)(ws + off); off += (size_t)KV_COLS * D_IN * 2;
  _Float16* wvh  = (_Float16*)(ws + off); off += (size_t)KV_COLS * D_IN * 2;
  _Float16* woh  = (_Float16*)(ws + off); off += (size_t)D_OUTV * D_OUTV * 2;
  _Float16* Qr   = (_Float16*)(ws + off); off += (size_t)BB * NUM_HEADS * SS * HEAD_DIM * 2;
  _Float16* Kr   = (_Float16*)(ws + off); off += (size_t)BB * NUM_KV * SS * HEAD_DIM * 2;
  _Float16* Vt   = (_Float16*)(ws + off); off += (size_t)BB * NUM_KV * HEAD_DIM * SS * 2;
  _Float16* ctxh = (_Float16*)(ws + off); off += (size_t)NTOK * D_OUTV * 2;

  {
    int n = NTOK * D_IN;
    cvt_f32_f16<<<(n + 255) / 256, 256, 0, stream>>>(x, xh, n);
  }
  {
    int n = D_IN * D_OUTV;
    cvt_w_t<<<(n + 255) / 256, 256, 0, stream>>>(wq, wqh, D_IN, D_OUTV);
    cvt_w_t<<<(n + 255) / 256, 256, 0, stream>>>(wo, woh, D_OUTV, D_OUTV);
  }
  {
    int n = D_IN * KV_COLS;
    cvt_w_t<<<(n + 255) / 256, 256, 0, stream>>>(wk, wkh, D_IN, KV_COLS);
    cvt_w_t<<<(n + 255) / 256, 256, 0, stream>>>(wv, wvh, D_IN, KV_COLS);
  }

  // QKV+RoPE: 128 row strips * 12 unit-groups
  qkv_rope_kernel<<<128 * 12, 128, 0, stream>>>(xh, wqh, wkh, wvh, cosT, sinT,
                                                Qr, Kr, Vt);

  // flash attention: B*H*(S/16) = 8192 waves
  attn_kernel<<<8192 / 4, 128, 0, stream>>>(Qr, Kr, Vt, ctxh);

  // output projection: 128 strips * 8 unit-groups
  out_proj_kernel<<<128 * 8, 128, 0, stream>>>(ctxh, woh, bo, out);
}
